// SelfAttention_51582557225176
// MI455X (gfx1250) — matrix operand order repaired and checked by reading, not compiled, not run on hardware
//
#include <hip/hip_runtime.h>
#include <math.h>

// ---------------- types / WMMA helpers ----------------
typedef _Float16 v16h __attribute__((ext_vector_type(16)));
typedef float    v8f  __attribute__((ext_vector_type(8)));
typedef _Float16 h2   __attribute__((ext_vector_type(2)));

union V16H { v16h v; h2 p[8]; _Float16 e[16]; };
union V8F  { v8f  v; float e[8]; };

#define NB 4
#define LSEQ 2048
#define DIM 1024
#define NH 16
#define EH 64

static __device__ __forceinline__ v8f wmma_f16(v16h a, v16h b, v8f c) {
  return __builtin_amdgcn_wmma_f32_16x16x32_f16(false, a, false, b, (short)0, c, false, false);
}

// A fragment, 16x32 f16. rowptr = &A[row=lane&15][k0]. ISA 7.12.2:
// lanes 0-15: V0..3 = K0..7, V4..7 = K16..23 ; lanes 16-31: +8.
static __device__ __forceinline__ v16h load_a(const _Float16* __restrict__ rowptr, int sub) {
  V16H f;
#pragma unroll
  for (int i = 0; i < 8; ++i) {
    const int g = i >> 2, j = i & 3;
    const int k = g * 16 + sub * 8 + j * 2;
    f.p[i] = *(const h2*)(rowptr + k);
  }
  return f.v;
}

// B fragment, 32x16 f16, read from a K-contiguous ("B-transposed") layout:
// colptr = &Bt[n=lane&15][k0]. lanes 0-15: V0..3 = K0..7, V4..7 = K16..23 ; lanes 16-31: +8.
static __device__ __forceinline__ v16h load_b(const _Float16* __restrict__ colptr, int sub) {
  V16H f;
#pragma unroll
  for (int i = 0; i < 8; ++i) {
    const int k = (i >> 2) * 16 + sub * 8 + (i & 3) * 2;
    f.p[i] = *(const h2*)(colptr + k);
  }
  return f.v;
}

// ---------------- kernel 0a: f32 -> f16 convert ----------------
__global__ void k_cvt(const float* __restrict__ in, _Float16* __restrict__ out, int n) {
  int i = blockIdx.x * blockDim.x + threadIdx.x;
  if (i < n) out[i] = (_Float16)in[i];
}

// ---------------- kernel 0b: weight transpose [h][d][e] -> [h][e][d] f16 ----------------
__global__ void k_wT(const float* __restrict__ in, _Float16* __restrict__ out) {
  int i = blockIdx.x * blockDim.x + threadIdx.x;   // NH*EH*DIM threads exactly
  const int d = i & (DIM - 1);
  const int r = i >> 10;
  const int e = r & (EH - 1);
  const int h = r >> 6;
  out[i] = (_Float16)in[((size_t)h * DIM + d) * EH + e];
}

// ---------------- kernel 1: QKV projection ----------------
// grid = NB*NH*(LSEQ/16) blocks, 128 threads (4 waves; wave w -> e-tile w*16)
__global__ __launch_bounds__(128) void k_qkv(
    const _Float16* __restrict__ xh,
    const _Float16* __restrict__ wqt, const _Float16* __restrict__ wkt,
    const _Float16* __restrict__ wvt,
    _Float16* __restrict__ Qh, _Float16* __restrict__ Kh, _Float16* __restrict__ Vt) {
  const int lane = threadIdx.x & 31;
  const int wv_  = threadIdx.x >> 5;
  const int m    = lane & 15;
  const int sub  = lane >> 4;

  int id = blockIdx.x;
  const int lt = id & 127; id >>= 7;
  const int h  = id & 15;
  const int b  = id >> 4;
  const int l0 = lt * 16, et = wv_ * 16;

  const _Float16* xrow = xh  + ((size_t)b * LSEQ + l0 + m) * DIM;
  const _Float16* qcol = wqt + ((size_t)h * EH + et + m) * DIM;
  const _Float16* kcol = wkt + ((size_t)h * EH + et + m) * DIM;
  const _Float16* vcol = wvt + ((size_t)h * EH + et + m) * DIM;

  v8f aq = {}, ak = {}, av = {};
  for (int kk = 0; kk < DIM; kk += 32) {
    v16h A = load_a(xrow + kk, sub);
    aq = wmma_f16(A, load_b(qcol + kk, sub), aq);
    ak = wmma_f16(A, load_b(kcol + kk, sub), ak);
    av = wmma_f16(A, load_b(vcol + kk, sub), av);
  }

  V8F q, k, v; q.v = aq; k.v = ak; v.v = av;
  const size_t hb = (size_t)(b * NH + h);
#pragma unroll
  for (int r = 0; r < 8; ++r) {
    const int row = r + 8 * sub;
    const size_t o = (hb * LSEQ + l0 + row) * EH + et + m;
    Qh[o] = (_Float16)q.e[r];
    Kh[o] = (_Float16)k.e[r];
    Vt[(hb * EH + et + m) * LSEQ + l0 + row] = (_Float16)v.e[r];  // V transposed [e][l]
  }
}

// ---------------- kernel 2: flash attention ----------------
// 8192 query tiles total; 4 waves/block, each wave owns one 16-row query tile.
__global__ __launch_bounds__(128) void k_attn(
    const _Float16* __restrict__ Qh, const _Float16* __restrict__ Kh,
    const _Float16* __restrict__ Vt, float* __restrict__ out) {
  __shared__ _Float16 Pt[4][16 * 32];

  const int lane = threadIdx.x & 31;
  const int w    = threadIdx.x >> 5;
  const int m    = lane & 15;
  const int sub  = lane >> 4;

  int id = blockIdx.x * 4 + w;
  const int qt = id & 127; id >>= 7;
  const int h  = id & 15;
  const int b  = id >> 4;
  const int q0 = qt * 16;
  const size_t hb = (size_t)(b * NH + h);

  const _Float16* Qrow = Qh + (hb * LSEQ + q0 + m) * EH;
  const v16h Qa0 = load_a(Qrow, sub);
  const v16h Qa1 = load_a(Qrow + 32, sub);

  float mrow[8], lrow[8];
#pragma unroll
  for (int r = 0; r < 8; ++r) { mrow[r] = -INFINITY; lrow[r] = 0.f; }
  V8F O0, O1, O2, O3;
  O0.v = (v8f){}; O1.v = (v8f){}; O2.v = (v8f){}; O3.v = (v8f){};

  _Float16* P = Pt[w];

  for (int key = 0; key < LSEQ; key += 32) {
    const _Float16* kc0 = Kh + (hb * LSEQ + key + m) * EH;  // keys key..key+15
    const _Float16* kc1 = kc0 + 16 * EH;                    // keys key+16..key+31
    __builtin_prefetch(kc0 + 32 * EH, 0, 1);

    v8f s0 = {}, s1 = {};
    s0 = wmma_f16(Qa0, load_b(kc0, sub), s0);
    s0 = wmma_f16(Qa1, load_b(kc0 + 32, sub), s0);
    s1 = wmma_f16(Qa0, load_b(kc1, sub), s1);
    s1 = wmma_f16(Qa1, load_b(kc1 + 32, sub), s1);

    V8F S0, S1; S0.v = s0; S1.v = s1;
    float sc[8];
#pragma unroll
    for (int r = 0; r < 8; ++r) {
      const float a = S0.e[r] * 0.125f;   // 1/sqrt(64)
      const float c = S1.e[r] * 0.125f;
      float mx = fmaxf(a, c);
      mx = fmaxf(mx, __shfl_xor(mx, 1));
      mx = fmaxf(mx, __shfl_xor(mx, 2));
      mx = fmaxf(mx, __shfl_xor(mx, 4));
      mx = fmaxf(mx, __shfl_xor(mx, 8));  // stays within 16-lane half-wave
      const float mnew  = fmaxf(mrow[r], mx);
      const float scale = __expf(mrow[r] - mnew);
      const float p0 = __expf(a - mnew);
      const float p1 = __expf(c - mnew);
      float sum = p0 + p1;
      sum += __shfl_xor(sum, 1);
      sum += __shfl_xor(sum, 2);
      sum += __shfl_xor(sum, 4);
      sum += __shfl_xor(sum, 8);
      lrow[r] = lrow[r] * scale + sum;
      mrow[r] = mnew;
      sc[r] = scale;
      const int row = r + 8 * sub;
      P[row * 32 + m]      = (_Float16)p0;
      P[row * 32 + m + 16] = (_Float16)p1;
    }
#pragma unroll
    for (int r = 0; r < 8; ++r) {
      O0.e[r] *= sc[r]; O1.e[r] *= sc[r];
      O2.e[r] *= sc[r]; O3.e[r] *= sc[r];
    }

    // P stores -> P loads: same-wave DS ops are processed in order; wait explicitly.
    asm volatile("s_wait_dscnt 0" ::: "memory");
    const v16h Pa = load_a(P + m * 32, sub);

    const _Float16* vc = Vt + (hb * EH + m) * LSEQ + key;   // Vt[n][k], n = t*16+m
    O0.v = wmma_f16(Pa, load_b(vc, sub), O0.v);
    O1.v = wmma_f16(Pa, load_b(vc + 16 * LSEQ, sub), O1.v);
    O2.v = wmma_f16(Pa, load_b(vc + 32 * LSEQ, sub), O2.v);
    O3.v = wmma_f16(Pa, load_b(vc + 48 * LSEQ, sub), O3.v);
  }

#pragma unroll
  for (int r = 0; r < 8; ++r) {
    const float inv = 1.f / lrow[r];
    const int row = r + 8 * sub;
    float* op = out + ((size_t)b * LSEQ + q0 + row) * DIM + h * EH + m;
    op[0]  = O0.e[r] * inv;
    op[16] = O1.e[r] * inv;
    op[32] = O2.e[r] * inv;
    op[48] = O3.e[r] * inv;
  }
}

// ---------------- launcher ----------------
extern "C" void kernel_launch(void* const* d_in, const int* in_sizes, int n_in,
                              void* d_out, int out_size, void* d_ws, size_t ws_size,
                              hipStream_t stream) {
  const float* x  = (const float*)d_in[0];
  const float* wk = (const float*)d_in[1];   // setup_inputs order: x, wk, wq, wv
  const float* wq = (const float*)d_in[2];
  const float* wv = (const float*)d_in[3];

  char* p = (char*)d_ws;
  const size_t xel = (size_t)NB * LSEQ * DIM;          // 8.4M
  const size_t wel = (size_t)NH * EH * DIM;            // 1.05M
  const size_t qel = (size_t)NB * NH * LSEQ * EH;      // 8.4M
  _Float16* xh  = (_Float16*)p; p += xel * 2;
  _Float16* wqt = (_Float16*)p; p += wel * 2;
  _Float16* wkt = (_Float16*)p; p += wel * 2;
  _Float16* wvt = (_Float16*)p; p += wel * 2;
  _Float16* Qh  = (_Float16*)p; p += qel * 2;
  _Float16* Kh  = (_Float16*)p; p += qel * 2;
  _Float16* Vt  = (_Float16*)p; p += qel * 2;

  k_cvt<<<(int)((xel + 255) / 256), 256, 0, stream>>>(x, xh, (int)xel);
  k_wT<<<(int)(wel / 256), 256, 0, stream>>>(wq, wqt);
  k_wT<<<(int)(wel / 256), 256, 0, stream>>>(wk, wkt);
  k_wT<<<(int)(wel / 256), 256, 0, stream>>>(wv, wvt);

  k_qkv<<<NB * NH * (LSEQ / 16), 128, 0, stream>>>(xh, wqt, wkt, wvt, Qh, Kh, Vt);
  k_attn<<<NB * NH * (LSEQ / 16) / 4, 128, 0, stream>>>(Qh, Kh, Vt, (float*)d_out);
}
